// LSTM_47347719471646
// MI455X (gfx1250) — compile-verified, matrix operand order checked by reading, not run on hardware
//
#include <hip/hip_runtime.h>
#include <hip/hip_bf16.h>

// ---------------------------------------------------------------------------
// LSTM encoder for MI455X (gfx1250): bf16 WMMA 16x16x32, persistent LDS state.
// ---------------------------------------------------------------------------

#define T_STEPS 19
#define NBATCH  8192
#define EMBD    64
#define HID     256
#define FFD     32
#define GATES   1024   // 4*HID
#define ROWS    64     // batch rows per workgroup
#define NT      256    // 8 wave32 per workgroup
#define SCALE_F 4.0f

// LDS strides (elements), padded to rotate banks while keeping 16B alignment
#define HSTR 264   // ushort stride for h buffers  (264*2B = 528B, 16B aligned)
#define ESTR 72    // ushort stride for emb        (72*2B  = 144B, 16B aligned)
#define CSTR 260   // float stride for c

typedef __attribute__((ext_vector_type(16))) __bf16 v16bf;
typedef __attribute__((ext_vector_type(8)))  float  v8f;

struct Pack32 { uint4 lo, hi; };

__device__ __forceinline__ v16bf pack_bf(uint4 lo, uint4 hi) {
  Pack32 p{lo, hi};
  return __builtin_bit_cast(v16bf, p);
}

__device__ __forceinline__ unsigned short f2bf(float x) {
  unsigned u = __builtin_bit_cast(unsigned, x);
  u += 0x7FFFu + ((u >> 16) & 1u);     // round-to-nearest-even
  return (unsigned short)(u >> 16);
}

__device__ __forceinline__ float sigmoidf_(float x) {
  return 1.0f / (1.0f + __expf(-x));
}
__device__ __forceinline__ float tanhf_(float x) {
  float e = __expf(-2.0f * __builtin_fabsf(x));
  float r = (1.0f - e) / (1.0f + e);
  return __builtin_copysignf(r, x);
}

// ---------------------------------------------------------------------------
// Prep: fp32 -> bf16 weight conversion (+ transpose of Wff to [FF][HID])
// ---------------------------------------------------------------------------
__global__ void lstm_prep_kernel(const float* __restrict__ Wih,
                                 const float* __restrict__ Whh,
                                 const float* __restrict__ Wff,
                                 unsigned short* __restrict__ wih_bf,
                                 unsigned short* __restrict__ whh_bf,
                                 unsigned short* __restrict__ wfft_bf) {
  int i = blockIdx.x * blockDim.x + threadIdx.x;
  if (i < GATES * EMBD) {
    wih_bf[i] = f2bf(Wih[i]);
  } else if (i < GATES * EMBD + GATES * HID) {
    int j = i - GATES * EMBD;
    whh_bf[j] = f2bf(Whh[j]);
  } else if (i < GATES * EMBD + GATES * HID + FFD * HID) {
    int j = i - (GATES * EMBD + GATES * HID);
    int row = j / HID;                 // ff output column 0..31
    int k   = j - row * HID;           // hidden index 0..255
    wfft_bf[j] = f2bf(Wff[(size_t)k * FFD + row]);   // transpose
  }
}

// ---------------------------------------------------------------------------
// Main persistent LSTM kernel. One WG = 64 batch rows for all 19 steps.
// ---------------------------------------------------------------------------
__global__ __launch_bounds__(NT) void lstm_main_kernel(
    const float* __restrict__ observed,              // [20][8192][2]
    const float* __restrict__ We,                    // [2][64]
    const float* __restrict__ be,                    // [64]
    const float* __restrict__ bih,                   // [1024]
    const float* __restrict__ bhh,                   // [1024]
    const float* __restrict__ bff,                   // [32]
    const unsigned short* __restrict__ wih_bf,       // [1024][64] bf16
    const unsigned short* __restrict__ whh_bf,       // [1024][256] bf16
    const unsigned short* __restrict__ wfft_bf,      // [32][256] bf16 (Wff^T)
    float* __restrict__ out)                         // [19][8192][32]
{
  __shared__ __align__(16) unsigned short sh_h[2][ROWS * HSTR]; // bf16, double buffer
  __shared__ __align__(16) float          sh_c[ROWS * CSTR];    // fp32 cell state
  __shared__ __align__(16) unsigned short sh_emb[ROWS * ESTR];  // bf16 embeddings
  __shared__ float sh_mask[ROWS];

  const int tid    = threadIdx.x;
  const int lane   = tid & 31;
  const int wv     = tid >> 5;           // 0..7
  const int col16  = lane & 15;          // N index within a 16x16 tile
  const int hihalf = (lane >> 4) & 1;    // lane half selects K sub-block
  const int n0     = blockIdx.x * ROWS;

  // h0 = c0 = 0
  for (int i = tid; i < ROWS * HSTR; i += NT) sh_h[0][i] = 0;
  for (int i = tid; i < ROWS * CSTR; i += NT) sh_c[i] = 0.0f;
  __syncthreads();

  for (int t = 0; t < T_STEPS; ++t) {
    const int rb = t & 1;       // read buffer for h
    const int wb = rb ^ 1;      // write buffer for h

    // ---- phase 1: velocity -> embedding (relu(vel*4 @ We + be)) + mask ----
    {
      const int row = tid >> 2;            // 0..63
      const int seg = tid & 3;             // 16 emb columns each
      const int n   = n0 + row;
      const size_t b1 = ((size_t)t       * NBATCH + n) * 2;
      const size_t b2 = ((size_t)(t + 1) * NBATCH + n) * 2;
      float o1x = observed[b1 + 0], o1y = observed[b1 + 1];
      float o2x = observed[b2 + 0], o2y = observed[b2 + 1];
      bool  m  = !((o1x != o1x) || (o2x != o2x));    // !(isnan | isnan)
      float vx = m ? (o2x - o1x) * SCALE_F : 0.0f;
      float vy = m ? (o2y - o1y) * SCALE_F : 0.0f;
      #pragma unroll
      for (int j = 0; j < 16; ++j) {
        int e = seg * 16 + j;
        float v = vx * We[e] + vy * We[EMBD + e] + be[e];
        v = v > 0.0f ? v : 0.0f;                      // relu
        sh_emb[row * ESTR + e] = f2bf(v);
      }
      if (seg == 0) sh_mask[row] = m ? 1.0f : 0.0f;
    }
    __syncthreads();

    // ---- phase 2: gates via WMMA + fused LSTM cell update ----
    // wave wv owns hidden column tiles hc = 2*wv, 2*wv+1 (cols wv*32..wv*32+31)
    for (int hc2 = 0; hc2 < 2; ++hc2) {
      const int hc   = wv * 2 + hc2;
      const int hcol = hc * 16 + col16;
      for (int r = 0; r < 4; ++r) {
        // accumulators for i/f/g/o gates, initialized with bih+bhh (per column)
        v8f acc0, acc1, acc2, acc3;
        {
          float b0 = bih[0 * HID + hcol] + bhh[0 * HID + hcol];
          float b1 = bih[1 * HID + hcol] + bhh[1 * HID + hcol];
          float b2 = bih[2 * HID + hcol] + bhh[2 * HID + hcol];
          float b3 = bih[3 * HID + hcol] + bhh[3 * HID + hcol];
          #pragma unroll
          for (int v = 0; v < 8; ++v) { acc0[v]=b0; acc1[v]=b1; acc2[v]=b2; acc3[v]=b3; }
        }
        // K over embedding: 64 = 2 k-tiles of 32
        #pragma unroll
        for (int kt = 0; kt < 2; ++kt) {
          const int kb = kt * 32 + hihalf * 8;
          const unsigned short* ea = &sh_emb[(r * 16 + col16) * ESTR + kb];
          v16bf A = pack_bf(*(const uint4*)ea, *(const uint4*)(ea + 16));
          #pragma unroll
          for (int g = 0; g < 4; ++g) {
            const unsigned short* wp =
                wih_bf + (size_t)(g * HID + hc * 16 + col16) * EMBD + kb;
            v16bf B = pack_bf(*(const uint4*)wp, *(const uint4*)(wp + 16));
            v8f& ac = g == 0 ? acc0 : g == 1 ? acc1 : g == 2 ? acc2 : acc3;
            ac = __builtin_amdgcn_wmma_f32_16x16x32_bf16(
                     false, A, false, B, (short)0, ac, false, false);
          }
        }
        // K over h: 256 = 8 k-tiles of 32
        #pragma unroll
        for (int kt = 0; kt < 8; ++kt) {
          const int kb = kt * 32 + hihalf * 8;
          const unsigned short* ha = &sh_h[rb][(r * 16 + col16) * HSTR + kb];
          v16bf A = pack_bf(*(const uint4*)ha, *(const uint4*)(ha + 16));
          #pragma unroll
          for (int g = 0; g < 4; ++g) {
            const unsigned short* wp =
                whh_bf + (size_t)(g * HID + hc * 16 + col16) * HID + kb;
            v16bf B = pack_bf(*(const uint4*)wp, *(const uint4*)(wp + 16));
            v8f& ac = g == 0 ? acc0 : g == 1 ? acc1 : g == 2 ? acc2 : acc3;
            ac = __builtin_amdgcn_wmma_f32_16x16x32_bf16(
                     false, A, false, B, (short)0, ac, false, false);
          }
        }
        // elementwise LSTM cell on the 16x16 tile (WMMA D layout:
        // lane l, vgpr v -> row = v + 8*(l>=16), col = l&15)
        #pragma unroll
        for (int v = 0; v < 8; ++v) {
          const int row  = r * 16 + v + hihalf * 8;
          const int colh = hc * 16 + col16;
          float m     = sh_mask[row];
          float i_    = sigmoidf_(acc0[v]);
          float f_    = sigmoidf_(acc1[v]);
          float g_    = tanhf_(acc2[v]);
          float o_    = sigmoidf_(acc3[v]);
          float c_old = sh_c[row * CSTR + colh];
          float c_new = f_ * c_old + i_ * g_;
          float h_new = o_ * tanhf_(c_new);
          unsigned short h_old = sh_h[rb][row * HSTR + colh];
          unsigned short h_bf  = f2bf(h_new);
          bool keep = (m != 0.0f);
          sh_c[row * CSTR + colh]     = keep ? c_new : c_old;
          sh_h[wb][row * HSTR + colh] = keep ? h_bf  : h_old;   // h = where(m, h_new, h)
        }
      }
    }
    __syncthreads();

    // ---- phase 3: feed-forward head: ff = where(m, h @ Wff + bff, NaN) ----
    // (masked rows output NaN regardless, so masked h in sh_h[wb] is fine)
    {
      const int r    = wv & 3;           // row tile
      const int fc   = wv >> 2;          // FF column tile (0/1)
      const int fcol = fc * 16 + col16;
      v8f acc;
      {
        float b = bff[fcol];
        #pragma unroll
        for (int v = 0; v < 8; ++v) acc[v] = b;
      }
      #pragma unroll
      for (int kt = 0; kt < 8; ++kt) {
        const int kb = kt * 32 + hihalf * 8;
        const unsigned short* ha = &sh_h[wb][(r * 16 + col16) * HSTR + kb];
        v16bf A = pack_bf(*(const uint4*)ha, *(const uint4*)(ha + 16));
        const unsigned short* wp = wfft_bf + (size_t)fcol * HID + kb;
        v16bf B = pack_bf(*(const uint4*)wp, *(const uint4*)(wp + 16));
        acc = __builtin_amdgcn_wmma_f32_16x16x32_bf16(
                  false, A, false, B, (short)0, acc, false, false);
      }
      const float fnan = __builtin_bit_cast(float, 0x7FC00000u);
      #pragma unroll
      for (int v = 0; v < 8; ++v) {
        const int row = r * 16 + v + hihalf * 8;
        float m   = sh_mask[row];
        float val = (m != 0.0f) ? acc[v] : fnan;
        out[((size_t)t * NBATCH + (n0 + row)) * FFD + fcol] = val;
      }
    }
    __syncthreads();
  }
}

// ---------------------------------------------------------------------------
extern "C" void kernel_launch(void* const* d_in, const int* in_sizes, int n_in,
                              void* d_out, int out_size, void* d_ws, size_t ws_size,
                              hipStream_t stream) {
  const float* observed = (const float*)d_in[0];
  // d_in[1] = goals (unused by reference output)
  const float* We  = (const float*)d_in[2];
  const float* be  = (const float*)d_in[3];
  const float* Wih = (const float*)d_in[4];
  const float* Whh = (const float*)d_in[5];
  const float* bih = (const float*)d_in[6];
  const float* bhh = (const float*)d_in[7];
  const float* Wff = (const float*)d_in[8];
  const float* bff = (const float*)d_in[9];
  // d_in[10] = batch_split (unused by reference output)

  unsigned short* wih_bf  = (unsigned short*)d_ws;               // 1024*64
  unsigned short* whh_bf  = wih_bf + GATES * EMBD;               // 1024*256
  unsigned short* wfft_bf = whh_bf + GATES * HID;                // 32*256

  const int total = GATES * EMBD + GATES * HID + FFD * HID;
  lstm_prep_kernel<<<(total + 255) / 256, 256, 0, stream>>>(
      Wih, Whh, Wff, wih_bf, whh_bf, wfft_bf);

  lstm_main_kernel<<<NBATCH / ROWS, NT, 0, stream>>>(
      observed, We, be, bih, bhh, bff, wih_bf, whh_bf, wfft_bf, (float*)d_out);
}
